// CausalMHSA_6193342841248
// MI455X (gfx1250) — matrix operand order repaired and checked by reading, not compiled, not run on hardware
//
#include <hip/hip_runtime.h>

// ---------------------------------------------------------------------------
// CDNA5 (gfx1250) causal MHSA + RoPE, bf16 WMMA everywhere, fp32 accumulate.
// ---------------------------------------------------------------------------

typedef __attribute__((ext_vector_type(8)))  __bf16 v8bf;
typedef __attribute__((ext_vector_type(16))) __bf16 v16bf;
typedef __attribute__((ext_vector_type(8)))  float  v8f;
typedef __attribute__((ext_vector_type(4)))  int    v4i;

#define AS1 __attribute__((address_space(1)))
#define AS3 __attribute__((address_space(3)))

#ifndef USE_ASYNC_LDS
#define USE_ASYNC_LDS 1
#endif
#if USE_ASYNC_LDS && __has_builtin(__builtin_amdgcn_global_load_async_to_lds_b128)
#define HAVE_ASYNC_LDS 1
#else
#define HAVE_ASYNC_LDS 0
#endif

constexpr int D_MODEL  = 1024;
constexpr int S_LEN    = 2048;
constexpr int BATCH    = 4;
constexpr int N_HEADS  = 16;
constexpr int HEAD_DIM = 64;
constexpr int M_TOTAL  = BATCH * S_LEN;   // 8192 rows in all big GEMMs
constexpr float SCALE  = 0.125f;          // 1/sqrt(64)

__device__ inline v8f v8f_zero() {
  v8f z;
#pragma unroll
  for (int i = 0; i < 8; ++i) z[i] = 0.0f;
  return z;
}

__device__ inline v16bf cat8(v8bf lo, v8bf hi) {
  return __builtin_shufflevector(lo, hi, 0,1,2,3,4,5,6,7,8,9,10,11,12,13,14,15);
}

// A-fragment (16x32 bf16, ISA 7.12.2): lane holds row m = lane%16;
// element e <-> k = (e/8)*16 + (lane/16)*8 + e%8  -> two contiguous 16B loads.
__device__ inline v16bf load_frag_a(const __bf16* row, int half) {
  v8bf lo = *reinterpret_cast<const v8bf*>(row + half * 8);
  v8bf hi = *reinterpret_cast<const v8bf*>(row + 16 + half * 8);
  return cat8(lo, hi);
}

// B-fragment (32x16 bf16): lane holds col n = lane%16; element e <-> k =
// (e/8)*16 + (lane/16)*8 + e%8. `col` points at k=0 of this lane's column, contiguous in k.
__device__ inline v16bf load_frag_b(const __bf16* col, int half) {
  v8bf lo = *reinterpret_cast<const v8bf*>(col + half * 8);
  v8bf hi = *reinterpret_cast<const v8bf*>(col + 16 + half * 8);
  return cat8(lo, hi);
}

__device__ inline v8f wmma_bf16(v16bf a, v16bf b, v8f c) {
  return __builtin_amdgcn_wmma_f32_16x16x32_bf16(false, a, false, b,
                                                 (short)0, c, false, false);
}

// ---------------------------------------------------------------------------
// fp32 -> bf16 conversion
// ---------------------------------------------------------------------------
__global__ void f32_to_bf16(const float* __restrict__ in,
                            __bf16* __restrict__ out, int n) {
  int i = blockIdx.x * blockDim.x + threadIdx.x;
  int stride = gridDim.x * blockDim.x;
  for (; i < n; i += stride) out[i] = (__bf16)in[i];
}

// ---------------------------------------------------------------------------
// Tiled WMMA GEMM: C[8192x1024] = A[8192x1024] * B[1024x1024]
// Block tile 128x128, 256 threads (8 waves); wave w owns a 32x64 sub-tile
// (2 row-tiles x 4 col-tiles). K staged in chunks of 64 through LDS.
// A tile staged with ASYNCcnt-tracked global_load_async_to_lds_b128.
// MODE 0: bf16 out   MODE 1: RoPE epilogue + bf16 out   MODE 2: fp32 out
// ---------------------------------------------------------------------------
template <int MODE>
__global__ __launch_bounds__(256)
void gemm_wmma(const __bf16* __restrict__ A, const __bf16* __restrict__ Bm,
               void* __restrict__ outp, const int* __restrict__ pos) {
  __shared__ __attribute__((aligned(16))) __bf16 As[128][64];
  __shared__ __attribute__((aligned(16))) __bf16 Bt[128][64];   // [n][k]

  const int tid  = threadIdx.x;
  const int lane = tid & 31;
  const int wv   = tid >> 5;          // 0..7
  const int half = lane >> 4;
  const int lm   = lane & 15;
  const int wrow = (wv >> 1) * 32;    // 0,32,64,96
  const int wcol = (wv & 1) * 64;     // 0,64
  const int row0 = blockIdx.x * 128;
  const int col0 = blockIdx.y * 128;

  v8f acc[2][4];
#pragma unroll
  for (int i = 0; i < 2; ++i)
#pragma unroll
    for (int j = 0; j < 4; ++j) acc[i][j] = v8f_zero();

  for (int kc = 0; kc < D_MODEL; kc += 64) {
    // ---- stage A tile 128x64 (each thread: 32 contiguous bf16) ----
    {
      const int r = tid >> 1;
      const int k = (tid & 1) * 32;
      const __bf16* src = &A[(size_t)(row0 + r) * D_MODEL + kc + k];
#if HAVE_ASYNC_LDS
#pragma unroll
      for (int i = 0; i < 4; ++i)
        __builtin_amdgcn_global_load_async_to_lds_b128(
            (AS1 v4i*)(src + i * 8), (AS3 v4i*)&As[r][k + i * 8], 0, 0);
#else
#pragma unroll
      for (int i = 0; i < 4; ++i)
        *reinterpret_cast<v8bf*>(&As[r][k + i * 8]) =
            *reinterpret_cast<const v8bf*>(src + i * 8);
#endif
      if (kc + 64 < D_MODEL)
        __builtin_prefetch(src + 64, 0, 1);   // global_prefetch_b8
    }
    // ---- stage B tile 64x128 transposed into Bt[n][k] ----
    {
      const int kk = tid >> 2;            // 0..63
      const int n  = (tid & 3) * 32;      // 0,32,64,96
      const __bf16* src = &Bm[(size_t)(kc + kk) * D_MODEL + col0 + n];
      v8bf bv[4];
#pragma unroll
      for (int v = 0; v < 4; ++v)
        bv[v] = *reinterpret_cast<const v8bf*>(src + v * 8);
#pragma unroll
      for (int v = 0; v < 4; ++v)
#pragma unroll
        for (int j = 0; j < 8; ++j)
          Bt[n + v * 8 + j][kk] = bv[v][j];
    }
#if HAVE_ASYNC_LDS
    asm volatile("s_wait_asynccnt 0" ::: "memory");
#endif
    __syncthreads();

#pragma unroll
    for (int ks = 0; ks < 64; ks += 32) {
      const v16bf a0 = load_frag_a(&As[wrow + lm][ks], half);
      const v16bf a1 = load_frag_a(&As[wrow + 16 + lm][ks], half);
#pragma unroll
      for (int nt = 0; nt < 4; ++nt) {
        const v16bf bfrag = load_frag_b(&Bt[wcol + nt * 16 + lm][ks], half);
        acc[0][nt] = wmma_bf16(a0, bfrag, acc[0][nt]);
        acc[1][nt] = wmma_bf16(a1, bfrag, acc[1][nt]);
      }
    }
    __syncthreads();
  }

  // ---- epilogue (C layout: VGPR r <-> row r+8*half, col = lane%16) ----
#pragma unroll
  for (int rt = 0; rt < 2; ++rt) {
#pragma unroll
    for (int nt = 0; nt < 4; ++nt) {
      const int col = col0 + wcol + nt * 16 + lm;
      const int d = col & (HEAD_DIM - 1);
      const float inv_freq = __powf(10000.0f, -(float)(d & ~1) / (float)HEAD_DIM);
#pragma unroll
      for (int r = 0; r < 8; ++r) {
        const int row = row0 + wrow + rt * 16 + r + 8 * half;
        float v = acc[rt][nt][r];
        if constexpr (MODE == 1) {
          const float p = (float)pos[row & (S_LEN - 1)];
          float sn, cs;
          __sincosf(p * inv_freq, &sn, &cs);
          const float partner = __shfl_xor(v, 1, 32);  // pair column d^1
          v = ((d & 1) == 0) ? (v * cs - partner * sn)   // rot_even
                             : (partner * sn + v * cs);  // rot_odd
        }
        if constexpr (MODE == 2) {
          ((float*)outp)[(size_t)row * D_MODEL + col] = v;
        } else {
          ((__bf16*)outp)[(size_t)row * D_MODEL + col] = (__bf16)v;
        }
      }
    }
  }
}

// ---------------------------------------------------------------------------
// Flash attention: one wave per 16-query tile; 4 waves / block.
// Per 32-key chunk: 4 WMMA for Q*K^T, online softmax, 4 WMMA for P*V.
// K^T B-fragments load straight from global (k-dim contiguous in K rows).
// V staged transposed in LDS; P bounced through LDS into A layout.
// ---------------------------------------------------------------------------
__global__ __launch_bounds__(128)
void flash_attn(const __bf16* __restrict__ Qb, const __bf16* __restrict__ Kb,
                const __bf16* __restrict__ Vb, __bf16* __restrict__ ctx) {
  __shared__ __attribute__((aligned(16))) __bf16 Ps[4][16][32];
  __shared__ __attribute__((aligned(16))) __bf16 Vt[4][64][32];  // [d][k]

  const int lane = threadIdx.x & 31;
  const int wv   = threadIdx.x >> 5;
  const int half = lane >> 4;
  const int lm   = lane & 15;
  const int qt   = blockIdx.x * 4 + wv;     // 0..127
  const int h    = blockIdx.y;
  const int b    = blockIdx.z;

  const size_t qrow0 = (size_t)b * S_LEN + qt * 16;
  const __bf16* qptr = Qb + (qrow0 + lm) * D_MODEL + h * HEAD_DIM;
  const v16bf qa0 = load_frag_a(qptr, half);        // d = 0..31
  const v16bf qa1 = load_frag_a(qptr + 32, half);   // d = 32..63

  v8f o[4];
#pragma unroll
  for (int i = 0; i < 4; ++i) o[i] = v8f_zero();
  float mrow[8], lrow[8];
#pragma unroll
  for (int r = 0; r < 8; ++r) { mrow[r] = -1e30f; lrow[r] = 0.0f; }

  const int kend = qt * 16 + 16;                    // keys needed (exclusive)
  for (int k0 = 0; k0 < kend; k0 += 32) {
    // ---- stage V chunk (32x64) transposed into Vt[wv][d][k] ----
    {
      const __bf16* vptr = Vb + ((size_t)b * S_LEN + k0 + lane) * D_MODEL + h * HEAD_DIM;
      v8bf vrow[8];
#pragma unroll
      for (int i = 0; i < 8; ++i)
        vrow[i] = *reinterpret_cast<const v8bf*>(vptr + i * 8);
#pragma unroll
      for (int i = 0; i < 8; ++i)
#pragma unroll
        for (int j = 0; j < 8; ++j)
          Vt[wv][i * 8 + j][lane] = vrow[i][j];
    }

    // ---- scores: S = Q (16x64) * K^T (64x32) ----
    const __bf16* kbase = Kb + ((size_t)b * S_LEN + k0) * D_MODEL + h * HEAD_DIM;
    v8f s0 = v8f_zero(), s1 = v8f_zero();
    {
      const __bf16* kcol0 = kbase + (size_t)lm * D_MODEL;          // keys k0+lm
      const __bf16* kcol1 = kbase + (size_t)(16 + lm) * D_MODEL;   // keys k0+16+lm
      s0 = wmma_bf16(qa0, load_frag_b(kcol0, half), s0);
      s0 = wmma_bf16(qa1, load_frag_b(kcol0 + 32, half), s0);
      s1 = wmma_bf16(qa0, load_frag_b(kcol1, half), s1);
      s1 = wmma_bf16(qa1, load_frag_b(kcol1 + 32, half), s1);
    }

    // ---- mask + scale + online softmax (rows r+8*half, per-half lanes) ----
#pragma unroll
    for (int r = 0; r < 8; ++r) {
      const int qrow = qt * 16 + r + 8 * half;
      const int key0 = k0 + lm;
      const int key1 = k0 + 16 + lm;
      float v0 = (key0 > qrow) ? -1e30f : s0[r] * SCALE;
      float v1 = (key1 > qrow) ? -1e30f : s1[r] * SCALE;

      float mx = fmaxf(v0, v1);
#pragma unroll
      for (int msk = 1; msk < 16; msk <<= 1)
        mx = fmaxf(mx, __shfl_xor(mx, msk, 32));
      const float mnew = fmaxf(mrow[r], mx);

      const float p0 = __expf(v0 - mnew);
      const float p1 = __expf(v1 - mnew);
      float ps = p0 + p1;
#pragma unroll
      for (int msk = 1; msk < 16; msk <<= 1)
        ps += __shfl_xor(ps, msk, 32);

      const float alpha = __expf(mrow[r] - mnew);
      lrow[r] = lrow[r] * alpha + ps;
      mrow[r] = mnew;
#pragma unroll
      for (int nt = 0; nt < 4; ++nt) o[nt][r] *= alpha;

      Ps[wv][r + 8 * half][lm]      = (__bf16)p0;
      Ps[wv][r + 8 * half][16 + lm] = (__bf16)p1;
    }

    // ---- O += P (16x32) * V (32x64) ----
    const v16bf pfrag = load_frag_a(&Ps[wv][lm][0], half);
#pragma unroll
    for (int nt = 0; nt < 4; ++nt) {
      const v16bf vfrag = load_frag_b(&Vt[wv][nt * 16 + lm][0], half);
      o[nt] = wmma_bf16(pfrag, vfrag, o[nt]);
    }
  }

  // ---- normalize + store ctx (bf16) ----
#pragma unroll
  for (int r = 0; r < 8; ++r) {
    const float inv = 1.0f / lrow[r];
    const size_t row = qrow0 + r + 8 * half;
#pragma unroll
    for (int nt = 0; nt < 4; ++nt)
      ctx[row * D_MODEL + h * HEAD_DIM + nt * 16 + lm] = (__bf16)(o[nt][r] * inv);
  }
}

// ---------------------------------------------------------------------------
// Launch
// ---------------------------------------------------------------------------
extern "C" void kernel_launch(void* const* d_in, const int* in_sizes, int n_in,
                              void* d_out, int out_size, void* d_ws, size_t ws_size,
                              hipStream_t stream) {
  const float* x   = (const float*)d_in[0];
  const int*   pos = (const int*)d_in[1];
  const float* Wq  = (const float*)d_in[2];
  const float* Wk  = (const float*)d_in[3];
  const float* Wv  = (const float*)d_in[4];
  const float* Wo  = (const float*)d_in[5];

  char* ws = (char*)d_ws;
  const size_t MB = 1u << 20;
  __bf16* xb  = (__bf16*)(ws + 0 * MB);    // 16 MB: x in bf16
  __bf16* wqb = (__bf16*)(ws + 16 * MB);   // 2 MB each
  __bf16* wkb = (__bf16*)(ws + 18 * MB);
  __bf16* wvb = (__bf16*)(ws + 20 * MB);
  __bf16* wob = (__bf16*)(ws + 22 * MB);
  __bf16* Qb  = (__bf16*)(ws + 24 * MB);   // 16 MB each (RoPE'd, bf16)
  __bf16* Kb  = (__bf16*)(ws + 40 * MB);
  __bf16* Vb  = (__bf16*)(ws + 56 * MB);
  __bf16* ctx = (__bf16*)(ws + 72 * MB);   // 16 MB

  const int nx = M_TOTAL * D_MODEL;
  const int nw = D_MODEL * D_MODEL;
  f32_to_bf16<<<1024, 256, 0, stream>>>(x,  xb,  nx);
  f32_to_bf16<<<512,  256, 0, stream>>>(Wq, wqb, nw);
  f32_to_bf16<<<512,  256, 0, stream>>>(Wk, wkb, nw);
  f32_to_bf16<<<512,  256, 0, stream>>>(Wv, wvb, nw);
  f32_to_bf16<<<512,  256, 0, stream>>>(Wo, wob, nw);

  dim3 ggrid(M_TOTAL / 128, D_MODEL / 128);  // (64, 8)
  gemm_wmma<1><<<ggrid, 256, 0, stream>>>(xb, wqb, Qb, pos);  // Q + RoPE
  gemm_wmma<1><<<ggrid, 256, 0, stream>>>(xb, wkb, Kb, pos);  // K + RoPE
  gemm_wmma<0><<<ggrid, 256, 0, stream>>>(xb, wvb, Vb, nullptr);

  flash_attn<<<dim3(S_LEN / 16 / 4, N_HEADS, BATCH), 128, 0, stream>>>(Qb, Kb, Vb, ctx);

  gemm_wmma<2><<<ggrid, 256, 0, stream>>>(ctx, wob, (float*)d_out, nullptr);
}
